// SelfAttention_50139448213843
// MI455X (gfx1250) — matrix operand order repaired and checked by reading, not compiled, not run on hardware
//
#include <hip/hip_runtime.h>

// ---------------------------------------------------------------------------
// Self-attention block for MI455X (gfx1250, wave32, WMMA + async-LDS).
//   B=4, C=256, H=W=64 (L=4096), heads=4, hd=64, groups=8.
// Compute-bound (~77 GFLOP vs ~100MB @ 23.3TB/s): all GEMM-shaped stages on
// v_wmma_f32_16x16x32_f16 (f32 accumulate); attention is flash-style.
// Tensor layouts in workspace are chosen so that every f16 LDS staging step
// is a pure 16B-chunk copy -> GLOBAL_LOAD_ASYNC_TO_LDS_B128 (ASYNCcnt path).
// ---------------------------------------------------------------------------

typedef _Float16 v16h __attribute__((ext_vector_type(16)));
typedef _Float16 v8h  __attribute__((ext_vector_type(8)));
typedef __fp16   v2y  __attribute__((ext_vector_type(2)));   // cvt_pkrtz result
typedef float    v8f  __attribute__((ext_vector_type(8)));

#define NB   4
#define NC   256
#define NL   4096
#define NOC  768
#define NH   4
#define HD   64

__device__ __forceinline__ v8f wmma16(v16h a, v16h b, v8f c) {
  return __builtin_amdgcn_wmma_f32_16x16x32_f16(
      false, a, false, b, (short)0, c, false, false);
}

// Async copy of one 16B chunk: global -> LDS, tracked by ASYNCcnt.
// LDS operand is the low 32 bits of the flat shared address (== LDS byte
// offset; LDS aperture keeps the offset in addr[31:0], ISA 10.2).
__device__ __forceinline__ void async_ld16(unsigned lds, const void* g) {
  asm volatile("global_load_async_to_lds_b128 %0, %1, off"
               :: "v"(lds), "v"((unsigned long long)(uintptr_t)g)
               : "memory");
}
__device__ __forceinline__ void wait_async0() {
  asm volatile("s_wait_asynccnt 0x0" ::: "memory");
}
__device__ __forceinline__ unsigned lds_lo32(const void* p) {
  return (unsigned)(uintptr_t)p;
}

// A-matrix fragment (ISA 7.12.2, 16-bit A 16x32): lane m = lane&15.
// lanes 0-15: v0..3 K0..7, v4..7 K16..23 ; lanes 16-31: +8.
__device__ __forceinline__ v16h frag_a(const _Float16* tile, int row,
                                       int stride, int koff) {
  const int lane = threadIdx.x & 31;
  const int k0 = koff + ((lane & 16) >> 1);
  const _Float16* p = tile + row * stride + k0;
  union { v16h v; v8h h[2]; } u;
  u.h[0] = *(const v8h*)(p);
  u.h[1] = *(const v8h*)(p + 16);
  return u.v;
}

// B-matrix fragment (16-bit B 32x16): lane n = lane&15; lanes 0-15 hold
// K0..7 and K16..23, lanes 16-31 hold K8..15 and K24..31.
__device__ __forceinline__ v16h frag_b(const _Float16* tile, int row,
                                       int stride, int koff) {
  const int lane = threadIdx.x & 31;
  const int k0 = koff + ((lane & 16) >> 1);
  const _Float16* p = tile + row * stride + k0;
  union { v16h v; v8h h[2]; } u;
  u.h[0] = *(const v8h*)(p);
  u.h[1] = *(const v8h*)(p + 16);
  return u.v;
}

// ---------------------------------------------------------------------------
// Kernel 1: GroupNorm (8 groups). One WG per (batch, group); 32ch x 4096px.
// ---------------------------------------------------------------------------
__global__ __launch_bounds__(256) void gn_kernel(const float* __restrict__ x,
                                                 const float* __restrict__ gamma,
                                                 const float* __restrict__ beta,
                                                 float* __restrict__ xn) {
  const int b = blockIdx.x >> 3, g = blockIdx.x & 7;
  const int tid = threadIdx.x;
  const size_t base = ((size_t)b * NC + g * 32) * NL;
  const int n = 32 * NL;
  float s = 0.f, ss = 0.f;
  for (int i = tid; i < n; i += 256) {
    float v = x[base + i];
    s += v; ss += v * v;
  }
  __shared__ float r0[256], r1[256];
  r0[tid] = s; r1[tid] = ss;
  __syncthreads();
  for (int off = 128; off > 0; off >>= 1) {
    if (tid < off) { r0[tid] += r0[tid + off]; r1[tid] += r1[tid + off]; }
    __syncthreads();
  }
  const float mean = r0[0] * (1.0f / 131072.0f);
  const float var  = r1[0] * (1.0f / 131072.0f) - mean * mean;
  const float inv  = rsqrtf(var + 1e-5f);
  for (int i = tid; i < n; i += 256) {
    int c = g * 32 + (i >> 12);
    float v = (x[base + i] - mean) * inv;
    xn[base + i] = v * gamma[c] + beta[c];
  }
}

// ---------------------------------------------------------------------------
// Kernel 2: QKV GEMM (768x4096) = W(768x256) * Xn(256x4096) per batch.
// WG tile 64(M)x128(N); 8 waves 2x4; wave 32x32 (2x2 WMMA). fp32 staged
// through packed v_cvt_pk_rtz_f16_f32. Q,K stored token-major [b,h,l,d];
// V stored [b,h,d,l] so attention can stage all tiles with async copies.
// ---------------------------------------------------------------------------
__global__ __launch_bounds__(256) void qkv_gemm(const float* __restrict__ W,
                                                const float* __restrict__ Xn,
                                                const float* __restrict__ bias,
                                                _Float16* __restrict__ Qt,
                                                _Float16* __restrict__ Kt,
                                                _Float16* __restrict__ Vt) {
  constexpr int TK = 32, AS = TK + 8;          // 40-half stride (80B)
  __shared__ __align__(16) _Float16 As[64 * AS];   // [m][k]
  __shared__ __align__(16) _Float16 Bs[128 * AS];  // [n][k] (transposed)
  const int b = blockIdx.z;
  const int m0 = blockIdx.y * 64;
  const int n0 = blockIdx.x * 128;
  const int tid = threadIdx.x;
  const int lane = tid & 31, wid = tid >> 5;
  const int wm = (wid & 1) * 32, wn = (wid >> 1) * 32;
  const float* Xb = Xn + (size_t)b * NC * NL;
  v8f c[2][2] = {};

  for (int k0 = 0; k0 < NC; k0 += TK) {
    __syncthreads();
    for (int e = tid; e < 64 * 16; e += 256) {           // A: float2 -> pk f16
      int r = e >> 4, c2 = (e & 15) * 2;
      float2 w = *(const float2*)&W[(m0 + r) * NC + k0 + c2];
      *(v2y*)&As[r * AS + c2] = __builtin_amdgcn_cvt_pkrtz(w.x, w.y);
    }
    for (int e = tid; e < 16 * 128; e += 256) {          // B: transpose+convert
      int r2 = e >> 7, cc = e & 127;
      float f0 = Xb[(size_t)(k0 + 2 * r2)     * NL + n0 + cc];
      float f1 = Xb[(size_t)(k0 + 2 * r2 + 1) * NL + n0 + cc];
      *(v2y*)&Bs[cc * AS + 2 * r2] = __builtin_amdgcn_cvt_pkrtz(f0, f1);
    }
    if (k0 + TK < NC)
      __builtin_prefetch(&Xb[(size_t)(k0 + TK) * NL + n0 + (tid & 127)], 0, 1);
    __syncthreads();
    v16h a0 = frag_a(As, wm + (lane & 15),      AS, 0);
    v16h a1 = frag_a(As, wm + 16 + (lane & 15), AS, 0);
    v16h b0 = frag_b(Bs, wn + (lane & 15),      AS, 0);
    v16h b1 = frag_b(Bs, wn + 16 + (lane & 15), AS, 0);
    c[0][0] = wmma16(a0, b0, c[0][0]);
    c[0][1] = wmma16(a0, b1, c[0][1]);
    c[1][0] = wmma16(a1, b0, c[1][0]);
    c[1][1] = wmma16(a1, b1, c[1][1]);
  }
  const int rbase = (lane >> 4) << 3;
  const int ncol  = lane & 15;
  const int sec   = m0 >> 8;                   // 0=q, 1=k, 2=v (64 | 256)
  for (int i = 0; i < 2; ++i)
    for (int j = 0; j < 2; ++j)
      for (int r = 0; r < 8; ++r) {
        int om = m0 + wm + i * 16 + rbase + r;
        int on = n0 + wn + j * 16 + ncol;
        _Float16 v = (_Float16)(c[i][j][r] + bias[om]);
        int h = (om >> 6) & 3, d = om & 63;
        if (sec == 0)
          Qt[((size_t)(b * NH + h) * NL + on) * HD + d] = v;      // [b,h,l,d]
        else if (sec == 1)
          Kt[((size_t)(b * NH + h) * NL + on) * HD + d] = v;      // [b,h,l,d]
        else
          Vt[((size_t)(b * NH + h) * HD + d) * NL + on] = v;      // [b,h,d,l]
      }
}

// ---------------------------------------------------------------------------
// Kernel 3: flash attention. WG = (128-query block, head, batch); 8 waves,
// each owns 16 queries. Q/K/V tiles staged with async global->LDS copies.
// Output stored token-major [b,l,c] so proj can async-stage its B tile.
// ---------------------------------------------------------------------------
__global__ __launch_bounds__(256) void attn_kernel(const _Float16* __restrict__ Qt,
                                                   const _Float16* __restrict__ Kt,
                                                   const _Float16* __restrict__ Vt,
                                                   _Float16* __restrict__ OutT) {
  constexpr int ST = HD + 8;                   // 72-half stride (144B)
  __shared__ __align__(16) _Float16 Qs[128 * ST];  // [q][d]
  __shared__ __align__(16) _Float16 Ks[64 * ST];   // [key][d]
  __shared__ __align__(16) _Float16 Vs[64 * ST];   // [d][key]
  __shared__ __align__(16) _Float16 Ps[128 * ST];  // per-wave [q16][key64]
  const int q0 = blockIdx.x * 128;
  const int h  = blockIdx.y;
  const int b  = blockIdx.z;
  const int tid = threadIdx.x;
  const int lane = tid & 31, wid = tid >> 5;
  const int mrow = lane & 15;
  const int rbase = (lane >> 4) << 3;

  const _Float16* Qg = Qt + ((size_t)(b * NH + h) * NL + q0) * HD;
  const _Float16* Kg = Kt + (size_t)(b * NH + h) * NL * HD;
  const _Float16* Vg = Vt + (size_t)(b * NH + h) * HD * NL;
  const unsigned qs0 = lds_lo32(Qs), ks0 = lds_lo32(Ks), vs0 = lds_lo32(Vs);

  // stage Q tile: 128 rows x 128B, pure copy -> async (1024 x 16B chunks)
  for (int cid = tid; cid < 128 * 8; cid += 256) {
    int qi = cid >> 3, ch = cid & 7;
    async_ld16(qs0 + qi * (ST * 2) + ch * 16, Qg + (size_t)qi * HD + ch * 8);
  }
  wait_async0();
  __syncthreads();
  v16h qa[2];
  qa[0] = frag_a(Qs, wid * 16 + mrow, ST, 0);
  qa[1] = frag_a(Qs, wid * 16 + mrow, ST, 32);

  float m_run[8], l_run[8];
  v8f o[4] = {};
  for (int r = 0; r < 8; ++r) { m_run[r] = -1e30f; l_run[r] = 0.f; }

  for (int key0 = 0; key0 < NL; key0 += 64) {
    __syncthreads();                           // prev block done with Ks/Vs/Ps
    for (int cid = tid; cid < 64 * 8; cid += 256) {  // K,V tiles: async copies
      int rr = cid >> 3, ch = cid & 7;
      async_ld16(ks0 + rr * (ST * 2) + ch * 16,
                 Kg + (size_t)(key0 + rr) * HD + ch * 8);
      async_ld16(vs0 + rr * (ST * 2) + ch * 16,
                 Vg + (size_t)rr * NL + key0 + ch * 8);
    }
    wait_async0();
    __syncthreads();

    v8f s[4] = {};
    for (int kk = 0; kk < 2; ++kk)             // S = Q^T K (K-dim = hd = 64)
      for (int j = 0; j < 4; ++j)
        s[j] = wmma16(qa[kk], frag_b(Ks, j * 16 + mrow, ST, kk * 32), s[j]);

    // online softmax; rows live in 16-lane halves of the C fragment
    float rmax[8], corr[8];
    for (int r = 0; r < 8; ++r) {
      for (int j = 0; j < 4; ++j) s[j][r] *= 0.125f;   // hd^-0.5
      float mx = fmaxf(fmaxf(s[0][r], s[1][r]), fmaxf(s[2][r], s[3][r]));
      for (int msk = 1; msk <= 8; msk <<= 1)
        mx = fmaxf(mx, __shfl_xor(mx, msk, 32));
      rmax[r] = mx;
    }
    for (int r = 0; r < 8; ++r) {
      float mnew = fmaxf(m_run[r], rmax[r]);
      float cr = __expf(m_run[r] - mnew);
      float sum = 0.f;
      for (int j = 0; j < 4; ++j) {
        float p = __expf(s[j][r] - mnew);
        s[j][r] = p; sum += p;
      }
      for (int msk = 1; msk <= 8; msk <<= 1)
        sum += __shfl_xor(sum, msk, 32);
      l_run[r] = l_run[r] * cr + sum;
      m_run[r] = mnew;
      corr[r] = cr;
    }
    for (int j = 0; j < 4; ++j)
      for (int r = 0; r < 8; ++r) o[j][r] *= corr[r];

    // transpose P (C layout) -> A layout via per-wave LDS strip
    _Float16* Pw = Ps + wid * 16 * ST;
    for (int j = 0; j < 4; ++j)
      for (int r = 0; r < 8; ++r)
        Pw[(rbase + r) * ST + j * 16 + mrow] = (_Float16)s[j][r];
    __syncthreads();

    for (int kk = 0; kk < 2; ++kk) {           // O += P V^T (K-dim = 64 keys)
      v16h pa = frag_a(Ps, wid * 16 + mrow, ST, kk * 32);
      for (int j = 0; j < 4; ++j)
        o[j] = wmma16(pa, frag_b(Vs, j * 16 + mrow, ST, kk * 32), o[j]);
    }
  }

  for (int r = 0; r < 8; ++r) l_run[r] = 1.0f / l_run[r];
  for (int j = 0; j < 4; ++j)
    for (int r = 0; r < 8; ++r) {
      int d  = j * 16 + mrow;
      int qi = q0 + wid * 16 + rbase + r;
      OutT[((size_t)b * NL + qi) * NC + h * HD + d] =      // token-major [b,l,c]
          (_Float16)(o[j][r] * l_run[r]);
    }
}

// ---------------------------------------------------------------------------
// Kernel 4: proj GEMM (256x256) * attn + bias + residual xn -> fp32 out.
// B tile ([token][chan], contiguous in k) staged with async copies.
// ---------------------------------------------------------------------------
__global__ __launch_bounds__(256) void proj_gemm(const float* __restrict__ W,
                                                 const _Float16* __restrict__ AtT,
                                                 const float* __restrict__ bias,
                                                 const float* __restrict__ Xn,
                                                 float* __restrict__ Out) {
  constexpr int TK = 32, AS = TK + 8;
  __shared__ __align__(16) _Float16 As[64 * AS];
  __shared__ __align__(16) _Float16 Bs[128 * AS];
  const int b = blockIdx.z;
  const int m0 = blockIdx.y * 64;
  const int n0 = blockIdx.x * 128;
  const int tid = threadIdx.x;
  const int lane = tid & 31, wid = tid >> 5;
  const int wm = (wid & 1) * 32, wn = (wid >> 1) * 32;
  const _Float16* Ab = AtT + (size_t)b * NL * NC;   // [token][chan]
  const unsigned bs0 = lds_lo32(Bs);
  v8f c[2][2] = {};

  for (int k0 = 0; k0 < NC; k0 += TK) {
    __syncthreads();
    for (int e = tid; e < 64 * 16; e += 256) {       // A: float2 -> pk f16
      int r = e >> 4, c2 = (e & 15) * 2;
      float2 w = *(const float2*)&W[(m0 + r) * NC + k0 + c2];
      *(v2y*)&As[r * AS + c2] = __builtin_amdgcn_cvt_pkrtz(w.x, w.y);
    }
    for (int cid = tid; cid < 128 * 4; cid += 256) { // B: async 16B chunks
      int n = cid >> 2, ch = cid & 3;
      async_ld16(bs0 + n * (AS * 2) + ch * 16,
                 Ab + (size_t)(n0 + n) * NC + k0 + ch * 8);
    }
    wait_async0();
    __syncthreads();
    v16h a0 = frag_a(As, wm + (lane & 15),      AS, 0);
    v16h a1 = frag_a(As, wm + 16 + (lane & 15), AS, 0);
    v16h b0 = frag_b(Bs, wn + (lane & 15),      AS, 0);
    v16h b1 = frag_b(Bs, wn + 16 + (lane & 15), AS, 0);
    c[0][0] = wmma16(a0, b0, c[0][0]);
    c[0][1] = wmma16(a0, b1, c[0][1]);
    c[1][0] = wmma16(a1, b0, c[1][0]);
    c[1][1] = wmma16(a1, b1, c[1][1]);
  }
  const int rbase = (lane >> 4) << 3;
  const int ncol  = lane & 15;
  for (int i = 0; i < 2; ++i)
    for (int j = 0; j < 2; ++j)
      for (int r = 0; r < 8; ++r) {
        int om = m0 + wm + i * 16 + rbase + r;
        int on = n0 + wn + j * 16 + ncol;
        size_t idx = ((size_t)b * NC + om) * NL + on;
        Out[idx] = c[i][j][r] + bias[om] + Xn[idx];
      }
}

// ---------------------------------------------------------------------------
extern "C" void kernel_launch(void* const* d_in, const int* in_sizes, int n_in,
                              void* d_out, int out_size, void* d_ws, size_t ws_size,
                              hipStream_t stream) {
  (void)in_sizes; (void)n_in; (void)out_size; (void)ws_size;
  const float* x      = (const float*)d_in[0];
  const float* qkv_w  = (const float*)d_in[1];
  const float* qkv_b  = (const float*)d_in[2];
  const float* proj_w = (const float*)d_in[3];
  const float* proj_b = (const float*)d_in[4];
  const float* gamma  = (const float*)d_in[5];
  const float* beta   = (const float*)d_in[6];
  float* out = (float*)d_out;

  // workspace layout (48 MB total)
  char* ws = (char*)d_ws;
  const size_t xn_bytes  = (size_t)NB * NC * NL * 4;          // 16.78 MB
  const size_t qkv_bytes = (size_t)NB * NH * NL * HD * 2;     //  8.39 MB each
  float*    xn   = (float*)ws;
  _Float16* Qt   = (_Float16*)(ws + xn_bytes);
  _Float16* Kt   = (_Float16*)(ws + xn_bytes + qkv_bytes);
  _Float16* Vt   = (_Float16*)(ws + xn_bytes + 2 * qkv_bytes);
  _Float16* attn = (_Float16*)(ws + xn_bytes + 3 * qkv_bytes);

  gn_kernel<<<dim3(NB * 8), 256, 0, stream>>>(x, gamma, beta, xn);
  qkv_gemm<<<dim3(NL / 128, NOC / 64, NB), 256, 0, stream>>>(qkv_w, xn, qkv_b,
                                                             Qt, Kt, Vt);
  attn_kernel<<<dim3(NL / 128, NH, NB), 256, 0, stream>>>(Qt, Kt, Vt, attn);
  proj_gemm<<<dim3(NL / 128, NC / 64, NB), 256, 0, stream>>>(proj_w, attn,
                                                             proj_b, xn, out);
}
